// FlatQuantQwen3MLP_70875550318899
// MI455X (gfx1250) — 8-bit matrix operand order repaired and checked by reading, compiled, not run on hardware
//
#include <hip/hip_runtime.h>
#include <hip/hip_bf16.h>
#include <stdint.h>


typedef __attribute__((ext_vector_type(8))) int   v8i;
typedef __attribute__((ext_vector_type(4))) int   v4i;
typedef __attribute__((ext_vector_type(2))) int   v2i;
typedef __attribute__((ext_vector_type(4))) unsigned int v4u;

#if defined(__has_builtin)
#if __has_builtin(__builtin_amdgcn_tensor_load_to_lds) && __has_builtin(__builtin_amdgcn_s_wait_tensorcnt)
#define HAVE_TDM 1
#endif
#endif
#ifndef HAVE_TDM
#define HAVE_TDM 0
#endif

__global__ __launch_bounds__(256) void gj_inverse_kernel(
    const float* s0, const float* s1, const float* s2, const float* s3,
    float* d0, float* d1, float* d2, float* d3) {
  __shared__ float M[128 * 128];
  const float* src;
  float* dst;
  int n;
  switch (blockIdx.x) {
    case 0: src = s0; dst = d0; n = 64; break;
    case 1: src = s1; dst = d1; n = 64; break;
    case 2: src = s2; dst = d2; n = 96; break;
    default: src = s3; dst = d3; n = 128; break;
  }
  const int tid = threadIdx.x;
  for (int i = tid; i < n * n; i += 256) M[i] = src[i];
  __syncthreads();
  for (int k = 0; k < n; ++k) {
    const float invp = 1.0f / M[k * n + k];
    __syncthreads();
    for (int j = tid; j < n; j += 256)
      M[k * n + j] = (j == k) ? invp : M[k * n + j] * invp;
    __syncthreads();
    for (int idx = tid; idx < n * n; idx += 256) {
      const int i = idx / n;
      const int j = idx - i * n;
      if (i != k && j != k) M[idx] -= M[i * n + k] * M[k * n + j];
    }
    __syncthreads();
    for (int i = tid; i < n; i += 256)
      if (i != k) M[i * n + k] = -M[i * n + k] * invp;
    __syncthreads();
  }
  for (int i = tid; i < n * n; i += 256) dst[i] = M[i];
}

__global__ __launch_bounds__(256) void kron_quant_kernel(
    const float* __restrict__ src, const float* __restrict__ diag,
    const float* __restrict__ M1, const float* __restrict__ M2,
    int dL, int dR, int inv_mode,
    signed char* __restrict__ qout, float* __restrict__ sout) {
  __shared__ float T1[96 * 128];
  __shared__ float line[128];
  __shared__ float red[256];
  const int tid = threadIdx.x;
  const int row = blockIdx.x;
  const int RS = dL * dR;
  const float* srow = src + (size_t)row * RS;

  for (int l = 0; l < dL; ++l) {
    for (int r = tid; r < dR; r += 256) {
      const float v = srow[l * dR + r];
      const float d = diag[l * dR + r];
      line[r] = inv_mode ? (v / d) : (v * d);
    }
    __syncthreads();
    for (int m = tid; m < dR; m += 256) {
      float acc = 0.0f;
      if (inv_mode) {
        const float* b = M2 + (size_t)m * dR;
        for (int r = 0; r < dR; ++r) acc += line[r] * b[r];
      } else {
        for (int r = 0; r < dR; ++r) acc += line[r] * M2[(size_t)r * dR + m];
      }
      T1[l * dR + m] = acc;
    }
    __syncthreads();
  }

  float lmax = 0.0f;
  for (int idx = tid; idx < RS; idx += 256) {
    const int k = idx / dR;
    const int m = idx - k * dR;
    float y = 0.0f;
    if (inv_mode) {
      const float* a = M1 + (size_t)k * dL;
      for (int l = 0; l < dL; ++l) y += a[l] * T1[l * dR + m];
    } else {
      for (int l = 0; l < dL; ++l) y += M1[(size_t)l * dL + k] * T1[l * dR + m];
    }
    lmax = fmaxf(lmax, fabsf(y));
  }
  red[tid] = lmax;
  __syncthreads();
  for (int s = 128; s > 0; s >>= 1) {
    if (tid < s) red[tid] = fmaxf(red[tid], red[tid + s]);
    __syncthreads();
  }
  const float scale = fmaxf(red[0] * (1.0f / 7.0f), 1e-8f);
  if (tid == 0) sout[row] = scale;
  const float inv_s = 1.0f / scale;

  signed char* qrow = qout + (size_t)row * RS;
  for (int idx = tid; idx < RS; idx += 256) {
    const int k = idx / dR;
    const int m = idx - k * dR;
    float y = 0.0f;
    if (inv_mode) {
      const float* a = M1 + (size_t)k * dL;
      for (int l = 0; l < dL; ++l) y += a[l] * T1[l * dR + m];
    } else {
      for (int l = 0; l < dL; ++l) y += M1[(size_t)l * dL + k] * T1[l * dR + m];
    }
    float q = rintf(y * inv_s);
    q = fminf(fmaxf(q, -7.0f), 7.0f);
    qrow[idx] = (signed char)(int)q;
  }
}

__device__ __forceinline__ v8i load_a_frag(const signed char* Abase, int lane) {
  const int row = lane & 15;
  const int hi = (lane >> 4) << 4;
  const signed char* p = Abase + row * 64 + hi;
  v8i a;
#pragma unroll
  for (int j = 0; j < 4; ++j) {
    const v2i c = *(const v2i*)(p + (j >> 1) * 32 + (j & 1) * 8);
    a[2 * j] = c[0];
    a[2 * j + 1] = c[1];
  }
  return a;
}

__device__ __forceinline__ v8i load_b_frag(const signed char* Bbase, int lane) {
  const int row = lane & 15;
  const int kh = (lane >> 4) << 4;
  const signed char* p = Bbase + row * 64 + kh;
  const v4i lo = *(const v4i*)(p);
  const v4i hv = *(const v4i*)(p + 32);
  v8i b;
  b[0] = lo[0]; b[1] = lo[1]; b[2] = lo[2]; b[3] = lo[3];
  b[4] = hv[0]; b[5] = hv[1]; b[6] = hv[2]; b[7] = hv[3];
  return b;
}

#if HAVE_TDM
__device__ __forceinline__ void tdm_load_b64x64(unsigned lds_addr,
                                                const signed char* gptr,
                                                unsigned K, unsigned nRows) {
  const unsigned long long ga = (unsigned long long)(uintptr_t)gptr;
  v4u g0;
  g0[0] = 1u;
  g0[1] = lds_addr;
  g0[2] = (unsigned)(ga & 0xFFFFFFFFu);
  g0[3] = (unsigned)((ga >> 32) & 0x1FFFFFFu) | (2u << 30);
  v8i g1;
  g1[0] = 0;
  g1[1] = (int)((K & 0xFFFFu) << 16);
  g1[2] = (int)((K >> 16) | ((nRows & 0xFFFFu) << 16));
  g1[3] = (int)((nRows >> 16) | (64u << 16));
  g1[4] = 64;
  g1[5] = (int)K;
  g1[6] = 0;
  g1[7] = 0;
  const v4i gz4 = {0, 0, 0, 0};
  const v8i gz8 = {0, 0, 0, 0, 0, 0, 0, 0};
  __builtin_amdgcn_tensor_load_to_lds(g0, g1, gz4, gz4, gz8, 0);
}
#endif

template <bool SWIGLU>
__global__ __launch_bounds__(256) void gemm_q4_kernel(
    const signed char* __restrict__ qa, const float* __restrict__ sa,
    const signed char* __restrict__ qb0, const float* __restrict__ sb0,
    const signed char* __restrict__ qb1, const float* __restrict__ sb1,
    int M, int N, int K, float* __restrict__ out) {
  constexpr int NW = SWIGLU ? 2 : 1;
  __shared__ signed char As[2][128 * 64];
  __shared__ signed char Bs[2][NW][64 * 64];

  const int tid = threadIdx.x;
  const int lane = tid & 31;
  const int wave = tid >> 5;
  const int wm = wave >> 1;
  const int wn = wave & 1;
  const int bn0 = blockIdx.x * 64;
  const int bm0 = blockIdx.y * 128;

  const v8i vzero = {0, 0, 0, 0, 0, 0, 0, 0};
  v8i acc[NW][2][2];
#pragma unroll
  for (int w = 0; w < NW; ++w)
#pragma unroll
    for (int i = 0; i < 2; ++i)
#pragma unroll
      for (int j = 0; j < 2; ++j) acc[w][i][j] = vzero;

  const int srow = tid >> 2;
  const int scol = (tid & 3) * 16;
  const signed char* gA = qa + (size_t)(bm0 + srow) * K + scol;
  const int nk = K >> 6;

  int4 aR0 = *(const int4*)(gA);
  int4 aR1 = *(const int4*)(gA + (size_t)64 * K);
#if HAVE_TDM
  if (wave == 0) {
    tdm_load_b64x64((unsigned)(uintptr_t)&Bs[0][0][0],
                    qb0 + (size_t)bn0 * K, (unsigned)K, 64u);
    if (SWIGLU)
      tdm_load_b64x64((unsigned)(uintptr_t)&Bs[0][NW - 1][0],
                      qb1 + (size_t)bn0 * K, (unsigned)K, 64u);
  }
#else
  const signed char* gB[NW];
  gB[0] = qb0 + (size_t)(bn0 + srow) * K + scol;
  if (SWIGLU) gB[1] = qb1 + (size_t)(bn0 + srow) * K + scol;
  int4 bR[NW];
#pragma unroll
  for (int w = 0; w < NW; ++w) bR[w] = *(const int4*)(gB[w]);
#endif
  *(int4*)(As[0] + srow * 64 + scol) = aR0;
  *(int4*)(As[0] + (srow + 64) * 64 + scol) = aR1;
#if HAVE_TDM
  if (wave == 0) __builtin_amdgcn_s_wait_tensorcnt(0);
#else
#pragma unroll
  for (int w = 0; w < NW; ++w) *(int4*)(&Bs[0][w][srow * 64 + scol]) = bR[w];
#endif
  __syncthreads();

  for (int kt = 0; kt < nk; ++kt) {
    const int cur = kt & 1;
    const int nxt = cur ^ 1;
    const bool has_next = (kt + 1 < nk);

    if (has_next) {
      const int kn = (kt + 1) << 6;
      aR0 = *(const int4*)(gA + kn);
      aR1 = *(const int4*)(gA + (size_t)64 * K + kn);
#if HAVE_TDM
      if (wave == 0) {
        tdm_load_b64x64((unsigned)(uintptr_t)&Bs[nxt][0][0],
                        qb0 + (size_t)bn0 * K + kn, (unsigned)K, 64u);
        if (SWIGLU)
          tdm_load_b64x64((unsigned)(uintptr_t)&Bs[nxt][NW - 1][0],
                          qb1 + (size_t)bn0 * K + kn, (unsigned)K, 64u);
      }
#else
#pragma unroll
      for (int w = 0; w < NW; ++w) bR[w] = *(const int4*)(gB[w] + kn);
#endif
      if (kt + 2 < nk) __builtin_prefetch(gA + kn + 64, 0, 1);
    }

    const v8i a0 = load_a_frag(As[cur] + (wm * 32) * 64, lane);
    const v8i a1 = load_a_frag(As[cur] + (wm * 32 + 16) * 64, lane);
#pragma unroll
    for (int w = 0; w < NW; ++w) {
      const v8i b0 = load_b_frag(&Bs[cur][w][(wn * 32) * 64], lane);
      const v8i b1 = load_b_frag(&Bs[cur][w][(wn * 32 + 16) * 64], lane);
      acc[w][0][0] = __builtin_amdgcn_wmma_i32_16x16x64_iu8(
          true, a0, true, b0, acc[w][0][0], false, false);
      acc[w][0][1] = __builtin_amdgcn_wmma_i32_16x16x64_iu8(
          true, a0, true, b1, acc[w][0][1], false, false);
      acc[w][1][0] = __builtin_amdgcn_wmma_i32_16x16x64_iu8(
          true, a1, true, b0, acc[w][1][0], false, false);
      acc[w][1][1] = __builtin_amdgcn_wmma_i32_16x16x64_iu8(
          true, a1, true, b1, acc[w][1][1], false, false);
    }

    if (has_next) {
      *(int4*)(As[nxt] + srow * 64 + scol) = aR0;
      *(int4*)(As[nxt] + (srow + 64) * 64 + scol) = aR1;
#if HAVE_TDM
      if (wave == 0) __builtin_amdgcn_s_wait_tensorcnt(0);
#else
#pragma unroll
      for (int w = 0; w < NW; ++w)
        *(int4*)(&Bs[nxt][w][srow * 64 + scol]) = bR[w];
#endif
    }
    __syncthreads();
  }

  const int hi8 = (lane >> 4) << 3;
  const int ln = lane & 15;
#pragma unroll
  for (int i = 0; i < 2; ++i) {
#pragma unroll
    for (int j = 0; j < 2; ++j) {
      const int gm0 = bm0 + wm * 32 + i * 16 + hi8;
      const int gn = bn0 + wn * 32 + j * 16 + ln;
      const float s0n = sb0[gn];
      if constexpr (SWIGLU) {
        const float s1n = sb1[gn];
#pragma unroll
        for (int v = 0; v < 8; ++v) {
          const int gm = gm0 + v;
          const float sax = sa[gm];
          const float g = (float)acc[0][i][j][v] * sax * s0n;
          const float u = (float)acc[NW - 1][i][j][v] * sax * s1n;
          out[(size_t)gm * N + gn] = (g / (1.0f + __expf(-g))) * u;
        }
      } else {
#pragma unroll
        for (int v = 0; v < 8; ++v) {
          const int gm = gm0 + v;
          out[(size_t)gm * N + gn] = (float)acc[0][i][j][v] * sa[gm] * s0n;
        }
      }
    }
  }
  (void)M;
}

extern "C" void kernel_launch(void* const* d_in, const int* in_sizes, int n_in,
                              void* d_out, int out_size, void* d_ws,
                              size_t ws_size, hipStream_t stream) {
  (void)in_sizes; (void)n_in; (void)out_size; (void)ws_size;
  const float* x       = (const float*)d_in[0];
  const float* w_gate  = (const float*)d_in[1];
  const float* w_up    = (const float*)d_in[2];
  const float* w_down  = (const float*)d_in[3];
  const float* ug_A    = (const float*)d_in[4];
  const float* ug_B    = (const float*)d_in[5];
  const float* ug_diag = (const float*)d_in[6];
  const float* dn_A    = (const float*)d_in[7];
  const float* dn_B    = (const float*)d_in[8];
  const float* dn_diag = (const float*)d_in[9];

  const int T = 2 * 2048;
  const int H = 4096;
  const int I = 12288;

  char* ws = (char*)d_ws;
  size_t off = 0;
  auto alloc = [&](size_t bytes) -> void* {
    void* p = ws + off;
    off += (bytes + 255) & ~(size_t)255;
    return p;
  };

  float* invAug = (float*)alloc(64 * 64 * 4);
  float* invBug = (float*)alloc(64 * 64 * 4);
  float* invAdn = (float*)alloc(96 * 96 * 4);
  float* invBdn = (float*)alloc(128 * 128 * 4);
  float* sx  = (float*)alloc((size_t)T * 4);
  float* sh  = (float*)alloc((size_t)T * 4);
  float* swg = (float*)alloc((size_t)I * 4);
  float* swu = (float*)alloc((size_t)I * 4);
  float* swd = (float*)alloc((size_t)H * 4);
  signed char* xq  = (signed char*)alloc((size_t)T * H);
  signed char* qwg = (signed char*)alloc((size_t)I * H);
  signed char* qwu = (signed char*)alloc((size_t)I * H);
  signed char* qwd = (signed char*)alloc((size_t)H * I);
  signed char* hq  = (signed char*)alloc((size_t)T * I);
  float* h = (float*)alloc((size_t)T * I * 4);

  gj_inverse_kernel<<<4, 256, 0, stream>>>(ug_A, ug_B, dn_A, dn_B,
                                           invAug, invBug, invAdn, invBdn);

  kron_quant_kernel<<<I, 256, 0, stream>>>(w_gate, ug_diag, invAug, invBug,
                                           64, 64, 1, qwg, swg);
  kron_quant_kernel<<<I, 256, 0, stream>>>(w_up, ug_diag, invAug, invBug,
                                           64, 64, 1, qwu, swu);
  kron_quant_kernel<<<H, 256, 0, stream>>>(w_down, dn_diag, invAdn, invBdn,
                                           96, 128, 1, qwd, swd);

  kron_quant_kernel<<<T, 256, 0, stream>>>(x, ug_diag, ug_A, ug_B,
                                           64, 64, 0, xq, sx);

  gemm_q4_kernel<true><<<dim3(I / 64, T / 128), 256, 0, stream>>>(
      xq, sx, qwg, swg, qwu, swu, T, I, H, h);

  kron_quant_kernel<<<T, 256, 0, stream>>>(h, dn_diag, dn_A, dn_B,
                                           96, 128, 0, hq, sh);

  gemm_q4_kernel<false><<<dim3(H / 64, T / 128), 256, 0, stream>>>(
      hq, sh, qwd, swd, nullptr, nullptr, T, H, I, (float*)d_out);
}
